// FullLinkedLayer_1477468750507
// MI455X (gfx1250) — matrix operand order repaired and checked by reading, not compiled, not run on hardware
//
#include <hip/hip_runtime.h>

// ---------------------------------------------------------------------------
// out[z,kl] = x[z,ij] * W[ij,kl] + bias[kl],  W = f0*f1*f2*f3 contraction.
// Pipeline: cvt_x (fp32->bf16) | pack_frags (f2/f3 -> WMMA B-frag layout)
//           build_w (fused tensor-network contraction, WMMA bf16)
//           gemm_out (8192x1024x4096 bf16 WMMA GEMM, double-buffered
//                     async global->LDS staging, ASYNCcnt-pipelined)
// ---------------------------------------------------------------------------

typedef __attribute__((ext_vector_type(16))) __bf16 v16bf;
typedef __attribute__((ext_vector_type(8)))  float  v8f;

union Frag {
  v16bf v;
  unsigned short s[16];
  unsigned int   u[8];
};

__device__ __forceinline__ unsigned short f2bf(float x) {
  unsigned int t = __float_as_uint(x);           // round-to-nearest-even bf16
  return (unsigned short)((t + 0x7FFFu + ((t >> 16) & 1u)) >> 16);
}

__device__ __forceinline__ v8f wmma_bf16(v16bf a, v16bf b, v8f c) {
  return __builtin_amdgcn_wmma_f32_16x16x32_bf16(false, a, false, b,
                                                 (short)0, c, false, false);
}

// ---------------------------------------------------------------- cvt_x ----
__global__ __launch_bounds__(256) void cvt_x(const float* __restrict__ x,
                                             unsigned short* __restrict__ xb) {
  const int i = blockIdx.x * 256 + threadIdx.x;   // grid exactly 8192*1024
  xb[i] = f2bf(x[i]);
}

// ----------------------------------------------------------- pack_frags ----
// Pre-swizzle f2 and f3 into WMMA B-fragment layout (bf16) so build_w's hot
// loop loads fragments as contiguous 32B chunks.
// f2p frag id = kc*64 + k  (kc: K-chunk of bd, k: output row of f2)
// f3p frag id = ce*4 + lt  (lt: 16-wide l tile), K>=16 zero padded.
__global__ __launch_bounds__(256) void pack_frags(const float* __restrict__ f2,
                                                  const float* __restrict__ f3,
                                                  unsigned short* __restrict__ f2p,
                                                  unsigned short* __restrict__ f3p) {
  const int gid  = blockIdx.x * 256 + threadIdx.x;  // 192*256 = 49152 = 1536 frags
  const int frag = gid >> 5;
  const int lane = gid & 31;
  const int m16  = lane & 15;
  const int hi16 = (lane & 16);                     // 0 or 16 (K lane-group)
  if (frag < 512) {                                 // f2: B[bd(256), (k,f)]
    const int kc = frag >> 6, k = frag & 63;
#pragma unroll
    for (int h = 0; h < 16; ++h) {
      const int bd = kc * 32 + (h & 8) * 2 + (hi16 >> 1) + (h & 7);  // interleaved (table) K split
      f2p[((size_t)frag * 32 + lane) * 16 + h] = f2bf(f2[(bd * 64 + k) * 16 + m16]);
    }
  } else {                                          // f3: B[f(16 pad 32), l]
    const int fid = frag - 512;                     // [0,1024)
    const int ce = fid >> 2, lt = fid & 3;
    const int c = ce >> 4, e = ce & 15;
#pragma unroll
    for (int h = 0; h < 16; ++h) {
      unsigned short v = 0;
      if (h < 8)                                    // rows K>=16 are zero
        v = f2bf(f3[((c * 16 + e) * 16 + (hi16 >> 1) + h) * 64 + lt * 16 + m16]);
      f3p[((size_t)fid * 32 + lane) * 16 + h] = v;
    }
  }
}

// -------------------------------------------------------------- build_w ----
// One workgroup per 16-row ij stripe of W (64 workgroups). Loop over the 256
// (c,e) bond slices; recompute T01_ce[16x256] in LDS, contract with f2 (K=256,
// 8 WMMAs per output k), bounce U through LDS to A-frag layout, contract with
// f3 (K=16 padded). Accumulate in registers; write Wt[kl][ij] bf16.
__global__ __launch_bounds__(256) void build_w(const float* __restrict__ f0,
                                               const float* __restrict__ f1,
                                               const unsigned short* __restrict__ f2p,
                                               const unsigned short* __restrict__ f3p,
                                               unsigned short* __restrict__ Wt) {
  __shared__ unsigned short Tt[16 * 256];   // T01 slice (bf16)
  __shared__ unsigned short Ubuf[8 * 256];  // per-wave D->A layout bounce

  const int tid   = threadIdx.x;
  const int wid   = tid >> 5;
  const int lane  = tid & 31;
  const int m16   = lane & 15;
  const int kgrp  = lane >> 4;             // 0/1
  const int kbase = kgrp ? 8 : 0;          // A-frag interleaved K base
  const int ij0   = blockIdx.x * 16;
  const int i_row = ij0 >> 5;              // constant across the 16-row tile
  const int jbase = ij0 & 31;
  const int bb    = tid >> 4;              // b index for T01 column
  const int dd    = tid & 15;              // d index for T01 column

  v8f acc[8][4];                           // [k within wave][l tile]
#pragma unroll
  for (int kk = 0; kk < 8; ++kk)
#pragma unroll
    for (int lt = 0; lt < 4; ++lt)
#pragma unroll
      for (int r = 0; r < 8; ++r) acc[kk][lt][r] = 0.0f;

  for (int ce = 0; ce < 256; ++ce) {
    const int c = ce >> 4, e = ce & 15;

    // ---- T01_ce[m=16][bd=256] = sum_a f0[i,a,b,c] * f1[a,j,d,e] (VALU) ----
    {
      float F0r[16];
#pragma unroll
      for (int a = 0; a < 16; ++a)
        F0r[a] = f0[((i_row * 16 + a) * 16 + bb) * 16 + c];
      for (int t = 0; t < 16; ++t) {
        float s = 0.0f;
#pragma unroll
        for (int a = 0; a < 16; ++a)
          s += F0r[a] * f1[((a * 32 + jbase + t) * 16 + dd) * 16 + e];
        Tt[t * 256 + tid] = f2bf(s);
      }
    }
    __syncthreads();                       // Tt ready for all waves

    // ---- A-fragments of T01 (8 K-chunks of 32) ----
    Frag at[8];
#pragma unroll
    for (int kc = 0; kc < 8; ++kc) {
#pragma unroll
      for (int t = 0; t < 4; ++t) {
        at[kc].u[t]     = *(const unsigned int*)&Tt[m16 * 256 + kc * 32 + kbase + 2 * t];
        at[kc].u[4 + t] = *(const unsigned int*)&Tt[m16 * 256 + kc * 32 + kbase + 16 + 2 * t];
      }
    }
    // ---- f3 B-fragments for this (c,e), 4 l-tiles ----
    Frag bf3[4];
#pragma unroll
    for (int lt = 0; lt < 4; ++lt) {
      const unsigned int* p =
          (const unsigned int*)(f3p + (((size_t)(ce * 4 + lt) * 32 + lane) << 4));
#pragma unroll
      for (int t = 0; t < 8; ++t) bf3[lt].u[t] = p[t];
    }
    __syncthreads();                       // all reads of Tt done -> reusable

    // ---- per output-k: U = T01 x f2 (K=256), then acc += U x f3 (K=16) ----
#pragma unroll
    for (int kk = 0; kk < 8; ++kk) {
      const int k = wid * 8 + kk;
      v8f U;
#pragma unroll
      for (int r = 0; r < 8; ++r) U[r] = 0.0f;
#pragma unroll
      for (int kc = 0; kc < 8; ++kc) {
        Frag bf2;
        const unsigned int* p =
            (const unsigned int*)(f2p + (((size_t)(kc * 64 + k) * 32 + lane) << 4));
#pragma unroll
        for (int t = 0; t < 8; ++t) bf2.u[t] = p[t];
        U = wmma_bf16(at[kc].v, bf2.v, U);
      }
      // D-layout -> A-layout bounce through per-wave LDS (DS ops are in-order)
#pragma unroll
      for (int r = 0; r < 8; ++r)
        Ubuf[wid * 256 + (r + kgrp * 8) * 16 + m16] = f2bf(U[r]);
      Frag ua;
#pragma unroll
      for (int t = 0; t < 4; ++t) {
        ua.u[t]     = *(const unsigned int*)&Ubuf[wid * 256 + m16 * 16 + kbase + 2 * t];
        ua.u[4 + t] = 0u;                  // K>=16 zero (pad to K=32)
      }
#pragma unroll
      for (int lt = 0; lt < 4; ++lt)
        acc[kk][lt] = wmma_bf16(ua.v, bf3[lt].v, acc[kk][lt]);
    }
  }

  // ---- store W transposed: Wt[kl][ij] (bf16), B-friendly for the GEMM ----
#pragma unroll
  for (int kk = 0; kk < 8; ++kk) {
    const int k = wid * 8 + kk;
#pragma unroll
    for (int lt = 0; lt < 4; ++lt)
#pragma unroll
      for (int r = 0; r < 8; ++r) {
        const int kl = k * 64 + lt * 16 + m16;
        Wt[(size_t)kl * 1024 + (ij0 + r + kgrp * 8)] = f2bf(acc[kk][lt][r]);
      }
  }
}

// ------------------------------------------------------------- gemm_out ----
// out[8192x4096] = Xb[8192x1024] * Wt^T + bias.  128x128 block tile, BK=32,
// 8 waves (2x4), wave tile 64x32.  Double-buffered async global->LDS staging:
// prefetch tile k+1 while computing tile k; async loads complete in order so
// `s_wait_asynccnt 4` drains the older tile's 4 copies while the newer 4
// remain in flight.

// Issue the 4 async-copy instructions (A chunk x2, B chunk x2) for one K-tile.
__device__ __forceinline__ void stage_tiles(const unsigned short* __restrict__ Xb,
                                            const unsigned short* __restrict__ Wt,
                                            unsigned short* AsB, unsigned short* BsB,
                                            int bm, int bn, int k0, int tid) {
#pragma unroll
  for (int p = 0; p < 2; ++p) {
    const int ca = tid + p * 256;
    const int row = ca >> 2, cc = ca & 3;
    {
      const unsigned short* g = Xb + (size_t)(bm + row) * 1024 + (k0 + cc * 8);
      unsigned int loff = (unsigned int)(uintptr_t)(AsB + ca * 8);
      asm volatile("global_load_async_to_lds_b128 %0, %1, off"
                   :: "v"(loff), "v"((unsigned long long)(uintptr_t)g)
                   : "memory");
    }
    {
      const unsigned short* g = Wt + (size_t)(bn + row) * 1024 + (k0 + cc * 8);
      unsigned int loff = (unsigned int)(uintptr_t)(BsB + ca * 8);
      asm volatile("global_load_async_to_lds_b128 %0, %1, off"
                   :: "v"(loff), "v"((unsigned long long)(uintptr_t)g)
                   : "memory");
    }
  }
}

// One BK=32 MMA step out of LDS (4 A-frags x 2 B-frags -> 8 WMMAs).
__device__ __forceinline__ void mma_step(const unsigned short* AsB,
                                         const unsigned short* BsB,
                                         v8f acc[4][2],
                                         int wm, int wn,
                                         int m16, int kgrp, int kbase) {
  Frag af[4], bq[2];
#pragma unroll
  for (int mf = 0; mf < 4; ++mf) {
    const int rowb = (wm + mf * 16 + m16) * 32;
#pragma unroll
    for (int t = 0; t < 4; ++t) {
      af[mf].u[t]     = *(const unsigned int*)&AsB[rowb + kbase + 2 * t];
      af[mf].u[4 + t] = *(const unsigned int*)&AsB[rowb + kbase + 16 + 2 * t];
    }
  }
#pragma unroll
  for (int nf = 0; nf < 2; ++nf) {
    const int rowb = (wn + nf * 16 + m16) * 32 + kgrp * 8;
#pragma unroll
    for (int t = 0; t < 8; ++t)
      bq[nf].u[t] = *(const unsigned int*)&BsB[rowb + 2 * t + (t & 4) * 2];
  }
#pragma unroll
  for (int mf = 0; mf < 4; ++mf)
#pragma unroll
    for (int nf = 0; nf < 2; ++nf)
      acc[mf][nf] = wmma_bf16(af[mf].v, bq[nf].v, acc[mf][nf]);
}

__global__ __launch_bounds__(256) void gemm_out(const unsigned short* __restrict__ Xb,
                                                const unsigned short* __restrict__ Wt,
                                                const float* __restrict__ bias,
                                                float* __restrict__ out) {
  __shared__ unsigned short As0[128 * 32], Bs0[128 * 32];   // ping
  __shared__ unsigned short As1[128 * 32], Bs1[128 * 32];   // pong

  const int tid   = threadIdx.x;
  const int lane  = tid & 31;
  const int wid   = tid >> 5;
  const int m16   = lane & 15;
  const int kgrp  = lane >> 4;
  const int kbase = kgrp ? 8 : 0;

  const int bm = blockIdx.x * 128;
  const int bn = blockIdx.y * 128;
  const int wm = (wid >> 2) * 64;
  const int wn = (wid & 3) * 32;

  v8f acc[4][2];
#pragma unroll
  for (int mf = 0; mf < 4; ++mf)
#pragma unroll
    for (int nf = 0; nf < 2; ++nf)
#pragma unroll
      for (int r = 0; r < 8; ++r) acc[mf][nf][r] = 0.0f;

  float biasr[2];
#pragma unroll
  for (int nf = 0; nf < 2; ++nf) biasr[nf] = bias[bn + wn + nf * 16 + m16];

  // prologue: prefetch first tile into ping
  stage_tiles(Xb, Wt, As0, Bs0, bm, bn, 0, tid);

  for (int k0 = 0; k0 < 1024; k0 += 64) {
    // prefetch k0+32 into pong, drain ping's 4 copies (in-order completion)
    stage_tiles(Xb, Wt, As1, Bs1, bm, bn, k0 + 32, tid);
    asm volatile("s_wait_asynccnt 0x4" ::: "memory");
    __syncthreads();
    mma_step(As0, Bs0, acc, wm, wn, m16, kgrp, kbase);
    __syncthreads();                       // all waves done reading ping

    // prefetch k0+64 into ping (if any), drain pong's copies
    if (k0 + 64 < 1024) {
      stage_tiles(Xb, Wt, As0, Bs0, bm, bn, k0 + 64, tid);
      asm volatile("s_wait_asynccnt 0x4" ::: "memory");
    } else {
      asm volatile("s_wait_asynccnt 0x0" ::: "memory");
    }
    __syncthreads();
    mma_step(As1, Bs1, acc, wm, wn, m16, kgrp, kbase);
    __syncthreads();                       // all waves done reading pong
  }

#pragma unroll
  for (int mf = 0; mf < 4; ++mf)
#pragma unroll
    for (int nf = 0; nf < 2; ++nf) {
      const int kl = bn + wn + nf * 16 + m16;
#pragma unroll
      for (int r = 0; r < 8; ++r) {
        const int z = bm + wm + mf * 16 + r + kgrp * 8;
        out[(size_t)z * 4096 + kl] = acc[mf][nf][r] + biasr[nf];
      }
    }
}

// --------------------------------------------------------------- launch ----
extern "C" void kernel_launch(void* const* d_in, const int* in_sizes, int n_in,
                              void* d_out, int out_size, void* d_ws, size_t ws_size,
                              hipStream_t stream) {
  const float* x    = (const float*)d_in[0];
  const float* f0   = (const float*)d_in[1];
  const float* f1   = (const float*)d_in[2];
  const float* f2   = (const float*)d_in[3];
  const float* f3   = (const float*)d_in[4];
  const float* bias = (const float*)d_in[5];
  float* out = (float*)d_out;

  unsigned short* xb  = (unsigned short*)d_ws;          // 8192*1024 bf16 = 16 MB
  unsigned short* Wt  = xb  + (size_t)8192 * 1024;      // 4096*1024 bf16 =  8 MB
  unsigned short* f2p = Wt  + (size_t)4096 * 1024;      // 512 frags  = 512 KB
  unsigned short* f3p = f2p + (size_t)512 * 32 * 16;    // 1024 frags =   1 MB

  cvt_x     <<<32768,        256, 0, stream>>>(x, xb);
  pack_frags<<<192,          256, 0, stream>>>(f2, f3, f2p, f3p);
  build_w   <<<64,           256, 0, stream>>>(f0, f1, f2p, f3p, Wt);
  gemm_out  <<<dim3(64, 32), 256, 0, stream>>>(xb, Wt, bias, out);
}
